// TimeSeriesPredictor_46772193853725
// MI455X (gfx1250) — matrix operand order repaired and checked by reading, not compiled, not run on hardware
//
#include <hip/hip_runtime.h>
#include <hip/hip_bf16.h>

typedef __attribute__((ext_vector_type(16))) _Float16 v16h;
typedef __attribute__((ext_vector_type(8)))  float    v8f;

__device__ __forceinline__ float sigf(float x) { return 1.f / (1.f + __expf(-x)); }
// tanh via hardware exp: tanh(x) = 2*sigmoid(2x) - 1
__device__ __forceinline__ float tanh_fast(float x) { return 2.f / (1.f + __expf(-2.f * x)) - 1.f; }

__device__ __forceinline__ v8f v8f_zero() {
    v8f z;
#pragma unroll
    for (int i = 0; i < 8; ++i) z[i] = 0.f;
    return z;
}

// ---------------------------------------------------------------------------
// Unguarded vector fragment loaders (tile fully in range).
// ---------------------------------------------------------------------------
__device__ __forceinline__ v16h load_a_vec(const float* __restrict__ arow, int k0, int lane)
{
    v16h a;
    const int kstart = k0 + ((lane < 16) ? 0 : 16);
    const float4* p0 = (const float4*)(arow + kstart);
    const float4* p1 = (const float4*)(arow + kstart + 8);
    float4 x0 = p0[0], x1 = p0[1], y0 = p1[0], y1 = p1[1];
    a[0] = (_Float16)x0.x;  a[1] = (_Float16)x0.y;  a[2] = (_Float16)x0.z;  a[3] = (_Float16)x0.w;
    a[4] = (_Float16)x1.x;  a[5] = (_Float16)x1.y;  a[6] = (_Float16)x1.z;  a[7] = (_Float16)x1.w;
    a[8] = (_Float16)y0.x;  a[9] = (_Float16)y0.y;  a[10] = (_Float16)y0.z; a[11] = (_Float16)y0.w;
    a[12] = (_Float16)y1.x; a[13] = (_Float16)y1.y; a[14] = (_Float16)y1.z; a[15] = (_Float16)y1.w;
    return a;
}

__device__ __forceinline__ v16h load_bt_vec(const float* __restrict__ brow, int k0, int lane)
{
    v16h b;
    const int kstart = k0 + ((lane < 16) ? 0 : 16);
    const float4* p = (const float4*)(brow + kstart);
    float4 x0 = p[0], x1 = p[1], x2 = p[2], x3 = p[3];
    b[0] = (_Float16)x0.x;  b[1] = (_Float16)x0.y;  b[2] = (_Float16)x0.z;  b[3] = (_Float16)x0.w;
    b[4] = (_Float16)x1.x;  b[5] = (_Float16)x1.y;  b[6] = (_Float16)x1.z;  b[7] = (_Float16)x1.w;
    b[8] = (_Float16)x2.x;  b[9] = (_Float16)x2.y;  b[10] = (_Float16)x2.z; b[11] = (_Float16)x2.w;
    b[12] = (_Float16)x3.x; b[13] = (_Float16)x3.y; b[14] = (_Float16)x3.z; b[15] = (_Float16)x3.w;
    return b;
}

__device__ __forceinline__ v16h load_bn_vec(const float* __restrict__ Bb, int ldb,
                                            int n, int k0, int lane)
{
    v16h b;
    const int kstart = k0 + ((lane < 16) ? 0 : 16);
    const float* p = Bb + (long long)kstart * ldb + n;
#pragma unroll
    for (int h = 0; h < 16; ++h) b[h] = (_Float16)p[(long long)h * ldb];
    return b;
}

// ---------------------------------------------------------------------------
// Masked (branch-free select) fragment loaders for edge tiles.
// ---------------------------------------------------------------------------
__device__ __forceinline__ v16h load_a_masked(const float* __restrict__ Ab, int lda,
                                              int M, int K, int m0, int k0, int lane)
{
    v16h a;
    const int m = m0 + (lane & 15);
    const bool mok = (m < M);
    const float* arow = Ab + (long long)(mok ? m : (M - 1)) * lda;
    const int kstart = k0 + ((lane < 16) ? 0 : 16);
#pragma unroll
    for (int h = 0; h < 16; ++h) {
        int k = kstart + h;
        bool ok = mok && (k < K);
        float v = arow[ok ? k : 0];
        a[h] = (_Float16)(ok ? v : 0.f);
    }
    return a;
}

__device__ __forceinline__ v16h load_b_masked(const float* __restrict__ Bb, int ldb,
                                              int N, int K, int n0, int k0, int transB,
                                              int lane)
{
    v16h b;
    const int n = n0 + (lane & 15);
    const bool nok = (n < N);
    const int nc = nok ? n : (N - 1);
    const int kstart = k0 + ((lane < 16) ? 0 : 16);
    if (transB) {
        const float* brow = Bb + (long long)nc * ldb;
#pragma unroll
        for (int h = 0; h < 16; ++h) {
            int k = kstart + h;
            bool ok = nok && (k < K);
            float v = brow[ok ? k : 0];
            b[h] = (_Float16)(ok ? v : 0.f);
        }
    } else {
#pragma unroll
        for (int h = 0; h < 16; ++h) {
            int k = kstart + h;
            bool ok = nok && (k < K);
            float v = Bb[(long long)(ok ? k : 0) * ldb + nc];
            b[h] = (_Float16)(ok ? v : 0.f);
        }
    }
    return b;
}

// ---------------------------------------------------------------------------
// Generic batched GEMM:  C[z] = alpha * A[z] @ op(B[z]) + bias   (f16 WMMA, f32 acc)
// Batch offset for z: (z/bdiv)*s?1 + (z%bdiv)*s?2.
// ---------------------------------------------------------------------------
__global__ void gemm_wmma_kernel(const float* __restrict__ A, const float* __restrict__ B,
                                 const float* __restrict__ bias, float* __restrict__ C,
                                 int M, int N, int K, int lda, int ldb, int ldc,
                                 int bdiv,
                                 long long sA1, long long sA2,
                                 long long sB1, long long sB2,
                                 long long sC1, long long sC2,
                                 int transB, int relu, float alpha)
{
    const int wave = threadIdx.x >> 5;
    const int lane = threadIdx.x & 31;
    const int tilesN = (N + 15) >> 4;
    const int tilesM = (M + 15) >> 4;
    const int tile = blockIdx.x * (blockDim.x >> 5) + wave;
    if (tile >= tilesM * tilesN) return;

    const int z  = blockIdx.y;
    const int z1 = z / bdiv, z2 = z % bdiv;
    const float* Ab = A + z1 * sA1 + z2 * sA2;
    const float* Bb = B + z1 * sB1 + z2 * sB2;
    float*       Cb = C + z1 * sC1 + z2 * sC2;

    const int mt = tile / tilesN, nt = tile % tilesN;
    const int m0 = mt << 4, n0 = nt << 4;

    v8f acc = v8f_zero();
    const int kfull = K & ~31;                       // multiple-of-32 prefix of K
    const bool inRange = (m0 + 16 <= M) && (n0 + 16 <= N);   // wave-uniform

    if (inRange) {
        const float* arow = Ab + (long long)(m0 + (lane & 15)) * lda;
        if (transB) {
            const float* brow = Bb + (long long)(n0 + (lane & 15)) * ldb;
            for (int k0 = 0; k0 < kfull; k0 += 32) {
                v16h a = load_a_vec(arow, k0, lane);
                v16h b = load_bt_vec(brow, k0, lane);
                acc = __builtin_amdgcn_wmma_f32_16x16x32_f16(false, a, false, b, (short)0,
                                                             acc, false, false);
            }
        } else {
            const int nB = n0 + (lane & 15);
            for (int k0 = 0; k0 < kfull; k0 += 32) {
                v16h a = load_a_vec(arow, k0, lane);
                v16h b = load_bn_vec(Bb, ldb, nB, k0, lane);
                acc = __builtin_amdgcn_wmma_f32_16x16x32_f16(false, a, false, b, (short)0,
                                                             acc, false, false);
            }
        }
        if (kfull < K) {   // one masked K-tail iteration
            v16h a = load_a_masked(Ab, lda, M, K, m0, kfull, lane);
            v16h b = load_b_masked(Bb, ldb, N, K, n0, kfull, transB, lane);
            acc = __builtin_amdgcn_wmma_f32_16x16x32_f16(false, a, false, b, (short)0,
                                                         acc, false, false);
        }
    } else {
        for (int k0 = 0; k0 < K; k0 += 32) {
            v16h a = load_a_masked(Ab, lda, M, K, m0, k0, lane);
            v16h b = load_b_masked(Bb, ldb, N, K, n0, k0, transB, lane);
            acc = __builtin_amdgcn_wmma_f32_16x16x32_f16(false, a, false, b, (short)0,
                                                         acc, false, false);
        }
    }

    const int nB = n0 + (lane & 15);
    const bool nok = (nB < N);
    const float bv = (bias != nullptr && nok) ? bias[nB] : 0.f;
    const int mh = (lane >> 4) << 3;
#pragma unroll
    for (int r = 0; r < 8; ++r) {
        int m = m0 + r + mh;
        if (m < M && nok) {
            float v = acc[r] * alpha + bv;
            if (relu) v = fmaxf(v, 0.f);
            Cb[(long long)m * ldc + nB] = v;
        }
    }
}

// ---------------------------------------------------------------------------
// Per-feature LSTM (input dim 1, H=64): 640 sequences, 16 per wave, state on chip.
// ---------------------------------------------------------------------------
__global__ void feat_lstm_kernel(const float* __restrict__ X,   // [16,128,40]
                                 const float* __restrict__ Wih, // [256]
                                 const float* __restrict__ bias,// [256]
                                 const float* __restrict__ Whh, // [256,64]
                                 float* __restrict__ Hout)      // [640,64]
{
    __shared__ __align__(32) _Float16 wpack[2 * 16 * 32 * 16];  // B-fragment packed Whh^T
    __shared__ _Float16 hbuf[8][16 * 64];
    __shared__ float    xt[8][16];

    const int tid = threadIdx.x;
    for (int idx = tid; idx < 2 * 16 * 32 * 16; idx += 256) {
        int h    = idx & 15;
        int lane = (idx >> 4) & 31;
        int nt   = (idx >> 9) & 15;
        int kt   = idx >> 13;
        int k = kt * 32 + ((lane < 16) ? h : h + 16);
        int n = nt * 16 + (lane & 15);
        wpack[idx] = (_Float16)Whh[n * 64 + k];
    }
    for (int idx = tid; idx < 8 * 16 * 64; idx += 256) ((_Float16*)hbuf)[idx] = (_Float16)0.f;
    __syncthreads();

    const int w = tid >> 5, lane = tid & 31;
    const int seqbase = blockIdx.x * 128 + w * 16;
    const int nloc = lane & 15;
    const int mh = (lane >> 4) * 8;

    v8f cst[4];
#pragma unroll
    for (int j = 0; j < 4; ++j) cst[j] = v8f_zero();

    for (int t = 0; t < 128; ++t) {
        if (lane < 16) {
            int s = seqbase + lane;
            xt[w][lane] = X[((s / 40) * 128 + t) * 40 + (s % 40)];
        }
        v16h a0, a1;
        {
            int m = lane & 15;
            int kb = (lane < 16) ? 0 : 16;
#pragma unroll
            for (int h = 0; h < 16; ++h) {
                int k = h + kb;
                a0[h] = hbuf[w][m * 64 + k];
                a1[h] = hbuf[w][m * 64 + 32 + k];
            }
        }
#pragma unroll
        for (int j = 0; j < 4; ++j) {
            v8f zg[4];
#pragma unroll
            for (int g = 0; g < 4; ++g) {
                int nt = g * 4 + j;
                int ng = nt * 16 + nloc;
                float wv = Wih[ng], bv = bias[ng];
                v8f acc;
#pragma unroll
                for (int r = 0; r < 8; ++r) acc[r] = wv * xt[w][r + mh] + bv;
                const v16h* b0 = (const v16h*)&wpack[((0 * 16 + nt) * 32 + lane) * 16];
                acc = __builtin_amdgcn_wmma_f32_16x16x32_f16(false, a0, false, *b0, (short)0,
                                                             acc, false, false);
                const v16h* b1 = (const v16h*)&wpack[((1 * 16 + nt) * 32 + lane) * 16];
                acc = __builtin_amdgcn_wmma_f32_16x16x32_f16(false, a1, false, *b1, (short)0,
                                                             acc, false, false);
                zg[g] = acc;
            }
#pragma unroll
            for (int r = 0; r < 8; ++r) {
                float ig = sigf(zg[0][r]);
                float fg = sigf(zg[1][r]);
                float gg = tanh_fast(zg[2][r]);
                float og = sigf(zg[3][r]);
                float cn = fg * cst[j][r] + ig * gg;
                cst[j][r] = cn;
                float hn = og * tanh_fast(cn);
                hbuf[w][(r + mh) * 64 + j * 16 + nloc] = (_Float16)hn;
            }
        }
    }
    for (int i = lane; i < 16 * 64; i += 32)
        Hout[(seqbase + i / 64) * 64 + (i % 64)] = (float)hbuf[w][i];
}

// ---------------------------------------------------------------------------
// Scalar LSTM with H=1 ("v" network).
// ---------------------------------------------------------------------------
__global__ void vlstm_kernel(const float* __restrict__ X, const float* __restrict__ Wih,
                             const float* __restrict__ Whh, const float* __restrict__ bias,
                             float* __restrict__ Vout)
{
    int idx = blockIdx.x * blockDim.x + threadIdx.x;
    if (idx >= 640) return;
    int b = idx / 40, f = idx % 40;
    float h = 0.f, c = 0.f;
    float wi0 = Wih[0], wi1 = Wih[1], wi2 = Wih[2], wi3 = Wih[3];
    float wh0 = Whh[0], wh1 = Whh[1], wh2 = Whh[2], wh3 = Whh[3];
    float b0 = bias[0], b1 = bias[1], b2 = bias[2], b3 = bias[3];
    for (int t = 0; t < 128; ++t) {
        float xv = X[(b * 128 + t) * 40 + f];
        float zi = wi0 * xv + b0 + wh0 * h;
        float zf = wi1 * xv + b1 + wh1 * h;
        float zg = wi2 * xv + b2 + wh2 * h;
        float zo = wi3 * xv + b3 + wh3 * h;
        c = sigf(zf) * c + sigf(zi) * tanh_fast(zg);
        h = sigf(zo) * tanh_fast(c);
    }
    Vout[idx] = h;
}

// ---------------------------------------------------------------------------
// Feature attention scores + softmax; writes Wsm [B,F,F] and b-output (transposed).
// ---------------------------------------------------------------------------
__global__ void scores_kernel(const float* __restrict__ Q, const float* __restrict__ K,
                              const float* __restrict__ V, float* __restrict__ Wsm,
                              float* __restrict__ bOut)
{
    int b = blockIdx.x / 40, i = blockIdx.x % 40;
    int j = threadIdx.x;  // 64
    __shared__ float smax[64], ssum[64];
    float s = -1e30f;
    if (j < 40) {
        const float* q = Q + (b * 40 + i) * 64;
        const float* k = K + (b * 40 + j) * 64;
        float d = 0.f;
        for (int h = 0; h < 64; ++h) d += q[h] * k[h];
        s = d * 0.125f * V[b * 40 + j];
    }
    smax[j] = s;
    __syncthreads();
    for (int off = 32; off; off >>= 1) {
        if (j < off) smax[j] = fmaxf(smax[j], smax[j + off]);
        __syncthreads();
    }
    float mx = smax[0];
    float e = (j < 40) ? __expf(s - mx) : 0.f;
    ssum[j] = e;
    __syncthreads();
    for (int off = 32; off; off >>= 1) {
        if (j < off) ssum[j] += ssum[j + off];
        __syncthreads();
    }
    float inv = 1.f / ssum[0];
    if (j < 40) {
        float p = e * inv;
        Wsm[(b * 40 + i) * 40 + j] = p;
        bOut[(b * 40 + j) * 40 + i] = p;   // b = Wsm.transpose(0,2,1)
    }
}

__global__ void copyx_kernel(const float* __restrict__ X, float* __restrict__ SRC)
{
    int idx = blockIdx.x * blockDim.x + threadIdx.x;
    if (idx >= 2048 * 40) return;
    int row = idx / 40, f = idx % 40;
    SRC[row * 80 + f] = X[idx];
}

__global__ void shuffle_kernel(const float* __restrict__ SRC, float* __restrict__ QS)
{
    int idx = blockIdx.x * blockDim.x + threadIdx.x;
    if (idx >= 2048 * 80) return;
    int row = idx / 80, col = idx % 80;
    int c = col / 5, g = col % 5;             // qs[..,c*5+g] = src[..,g*16+c]
    QS[idx] = SRC[row * 80 + g * 16 + c];
}

// wave-per-row softmax over 128 columns, in place
__global__ void softmax128_kernel(float* __restrict__ data, int rows)
{
    int w = threadIdx.x >> 5, lane = threadIdx.x & 31;
    int row = blockIdx.x * (blockDim.x >> 5) + w;
    if (row >= rows) return;
    float* p = data + (long long)row * 128;
    float v[4];
    float mx = -1e30f;
#pragma unroll
    for (int i = 0; i < 4; ++i) { v[i] = p[lane + 32 * i]; mx = fmaxf(mx, v[i]); }
    for (int off = 16; off; off >>= 1) mx = fmaxf(mx, __shfl_xor(mx, off, 32));
    float sum = 0.f;
#pragma unroll
    for (int i = 0; i < 4; ++i) { v[i] = __expf(v[i] - mx); sum += v[i]; }
    for (int off = 16; off; off >>= 1) sum += __shfl_xor(sum, off, 32);
    float inv = 1.f / sum;
#pragma unroll
    for (int i = 0; i < 4; ++i) p[lane + 32 * i] = v[i] * inv;
}

// src = LayerNorm(src + res) over 80 columns; wave per row
__global__ void add_ln_kernel(float* __restrict__ src, const float* __restrict__ res,
                              const float* __restrict__ g, const float* __restrict__ bb,
                              int rows)
{
    int w = threadIdx.x >> 5, lane = threadIdx.x & 31;
    int row = blockIdx.x * (blockDim.x >> 5) + w;
    if (row >= rows) return;
    float* s = src + (long long)row * 80;
    const float* r = res + (long long)row * 80;
    float v[3];
    float sum = 0.f;
#pragma unroll
    for (int i = 0; i < 3; ++i) {
        int j = lane + 32 * i;
        v[i] = (j < 80) ? (s[j] + r[j]) : 0.f;
        sum += v[i];
    }
    for (int off = 16; off; off >>= 1) sum += __shfl_xor(sum, off, 32);
    float mean = sum * (1.f / 80.f);
    float vs = 0.f;
#pragma unroll
    for (int i = 0; i < 3; ++i) {
        int j = lane + 32 * i;
        float d = (j < 80) ? (v[i] - mean) : 0.f;
        vs += d * d;
    }
    for (int off = 16; off; off >>= 1) vs += __shfl_xor(vs, off, 32);
    float rstd = rsqrtf(vs * (1.f / 80.f) + 1e-5f);
#pragma unroll
    for (int i = 0; i < 3; ++i) {
        int j = lane + 32 * i;
        if (j < 80) s[j] = (v[i] - mean) * rstd * g[j] + bb[j];
    }
}

// ---------------------------------------------------------------------------
// Bidirectional LSTM recurrence: batch=16 in one wave, fwd/bwd as two blocks.
// ---------------------------------------------------------------------------
__global__ void bilstm_rec_kernel(const float* __restrict__ gxF, const float* __restrict__ gxB,
                                  const float* __restrict__ WhhF, const float* __restrict__ WhhB,
                                  float* __restrict__ out)
{
    __shared__ __align__(32) _Float16 wpack[2 * 16 * 32 * 16];
    __shared__ _Float16 hbuf[16 * 64];
    const int dirb = blockIdx.x;
    const float* gx  = dirb ? gxB : gxF;
    const float* Whh = dirb ? WhhB : WhhF;
    const int tid = threadIdx.x;
    for (int idx = tid; idx < 2 * 16 * 32 * 16; idx += 256) {
        int h = idx & 15;
        int lane = (idx >> 4) & 31;
        int nt = (idx >> 9) & 15;
        int kt = idx >> 13;
        int k = kt * 32 + ((lane < 16) ? h : h + 16);
        int n = nt * 16 + (lane & 15);
        wpack[idx] = (_Float16)Whh[n * 64 + k];
    }
    for (int idx = tid; idx < 16 * 64; idx += 256) hbuf[idx] = (_Float16)0.f;
    __syncthreads();
    if (tid >= 32) return;
    const int lane = tid;
    const int nloc = lane & 15;
    const int mh = (lane >> 4) * 8;

    v8f cst[4];
#pragma unroll
    for (int j = 0; j < 4; ++j) cst[j] = v8f_zero();

    for (int step = 0; step < 128; ++step) {
        int t = dirb ? (127 - step) : step;
        v16h a0, a1;
        {
            int m = lane & 15;
            int kb = (lane < 16) ? 0 : 16;
#pragma unroll
            for (int h = 0; h < 16; ++h) {
                int k = h + kb;
                a0[h] = hbuf[m * 64 + k];
                a1[h] = hbuf[m * 64 + 32 + k];
            }
        }
#pragma unroll
        for (int j = 0; j < 4; ++j) {
            v8f zg[4];
#pragma unroll
            for (int g = 0; g < 4; ++g) {
                int nt = g * 4 + j;
                v8f acc;
#pragma unroll
                for (int r = 0; r < 8; ++r) {
                    int m = r + mh;
                    acc[r] = gx[((long long)m * 128 + t) * 256 + nt * 16 + nloc];
                }
                const v16h* b0 = (const v16h*)&wpack[((0 * 16 + nt) * 32 + lane) * 16];
                acc = __builtin_amdgcn_wmma_f32_16x16x32_f16(false, a0, false, *b0, (short)0,
                                                             acc, false, false);
                const v16h* b1 = (const v16h*)&wpack[((1 * 16 + nt) * 32 + lane) * 16];
                acc = __builtin_amdgcn_wmma_f32_16x16x32_f16(false, a1, false, *b1, (short)0,
                                                             acc, false, false);
                zg[g] = acc;
            }
#pragma unroll
            for (int r = 0; r < 8; ++r) {
                float ig = sigf(zg[0][r]);
                float fg = sigf(zg[1][r]);
                float gv = tanh_fast(zg[2][r]);
                float og = sigf(zg[3][r]);
                float cn = fg * cst[j][r] + ig * gv;
                cst[j][r] = cn;
                float hn = og * tanh_fast(cn);
                hbuf[(r + mh) * 64 + j * 16 + nloc] = (_Float16)hn;
                out[(((long long)(r + mh)) * 128 + t) * 128 + dirb * 64 + j * 16 + nloc] = hn;
            }
        }
    }
}

// final head: relu(out1[:,127,:]) -> relu(fc1) -> fc2
__global__ void head_kernel(const float* __restrict__ out1, const float* __restrict__ fc1W,
                            const float* __restrict__ fc1b, const float* __restrict__ fc2W,
                            const float* __restrict__ fc2b, float* __restrict__ out)
{
    int b = blockIdx.x;
    __shared__ float last[128], h1[64], red[64];
    int tid = threadIdx.x;  // 128
    last[tid] = fmaxf(out1[((long long)b * 128 + 127) * 128 + tid], 0.f);
    __syncthreads();
    if (tid < 64) {
        float a = fc1b[tid];
        for (int k = 0; k < 128; ++k) a += last[k] * fc1W[tid * 128 + k];
        h1[tid] = fmaxf(a, 0.f);
    }
    __syncthreads();
    if (tid < 64) red[tid] = h1[tid] * fc2W[tid];
    __syncthreads();
    for (int off = 32; off; off >>= 1) {
        if (tid < off) red[tid] += red[tid + off];
        __syncthreads();
    }
    if (tid == 0) out[b] = red[0] + fc2b[0];
}

// ---------------------------------------------------------------------------
static inline void launch_gemm(hipStream_t st, const float* A, const float* B,
                               const float* bias, float* C, int M, int N, int K,
                               int lda, int ldb, int ldc, int batch, int bdiv,
                               long long sA1, long long sA2, long long sB1, long long sB2,
                               long long sC1, long long sC2, int transB, int relu,
                               float alpha)
{
    int tiles = ((M + 15) / 16) * ((N + 15) / 16);
    dim3 grid((tiles + 3) / 4, batch);
    gemm_wmma_kernel<<<grid, 128, 0, st>>>(A, B, bias, C, M, N, K, lda, ldb, ldc, bdiv,
                                           sA1, sA2, sB1, sB2, sC1, sC2, transB, relu, alpha);
}

extern "C" void kernel_launch(void* const* d_in, const int* in_sizes, int n_in,
                              void* d_out, int out_size, void* d_ws, size_t ws_size,
                              hipStream_t stream)
{
    const float* x     = (const float*)d_in[0];
    const float* qWih  = (const float*)d_in[1];
    const float* qWhh  = (const float*)d_in[2];
    const float* qb    = (const float*)d_in[3];
    const float* kWih  = (const float*)d_in[4];
    const float* kWhh  = (const float*)d_in[5];
    const float* kb    = (const float*)d_in[6];
    const float* vWih  = (const float*)d_in[7];
    const float* vWhh  = (const float*)d_in[8];
    const float* vb    = (const float*)d_in[9];
    const float* tWq   = (const float*)d_in[10];
    const float* tbq   = (const float*)d_in[11];
    const float* tWk   = (const float*)d_in[12];
    const float* tbk   = (const float*)d_in[13];
    const float* tWv   = (const float*)d_in[14];
    const float* tbv   = (const float*)d_in[15];
    const float* tWo   = (const float*)d_in[16];
    const float* tbo   = (const float*)d_in[17];
    const float* tW1   = (const float*)d_in[18];
    const float* tb1   = (const float*)d_in[19];
    const float* tW2   = (const float*)d_in[20];
    const float* tb2   = (const float*)d_in[21];
    const float* tlng  = (const float*)d_in[22];
    const float* tlnb  = (const float*)d_in[23];
    const float* l0fWih = (const float*)d_in[24];
    const float* l0fWhh = (const float*)d_in[25];
    const float* l0fb   = (const float*)d_in[26];
    const float* l0bWih = (const float*)d_in[27];
    const float* l0bWhh = (const float*)d_in[28];
    const float* l0bb   = (const float*)d_in[29];
    const float* l1fWih = (const float*)d_in[30];
    const float* l1fWhh = (const float*)d_in[31];
    const float* l1fb   = (const float*)d_in[32];
    const float* l1bWih = (const float*)d_in[33];
    const float* l1bWhh = (const float*)d_in[34];
    const float* l1bb   = (const float*)d_in[35];
    const float* fc1W  = (const float*)d_in[36];
    const float* fc1b  = (const float*)d_in[37];
    const float* fc2W  = (const float*)d_in[38];
    const float* fc2b  = (const float*)d_in[39];

    float* outP = (float*)d_out;       // [16]
    float* bOut = outP + 16;           // [16,40,40]
    float* cOut = bOut + 25600;        // [2,16,5,128,128]

    float* ws = (float*)d_ws;
    size_t off = 0;
    auto alloc = [&](size_t n) { float* p = ws + off; off += n; return p; };
    float* Qh  = alloc(640 * 64);
    float* Kh  = alloc(640 * 64);
    float* Vv  = alloc(640);
    float* WSM = alloc(16 * 40 * 40);
    float* SRC = alloc(2048 * 80);
    float* QS  = alloc(2048 * 80);
    float* QB  = alloc(2048 * 80);
    float* KB  = alloc(2048 * 80);
    float* VB  = alloc(2048 * 80);
    float* OB  = alloc(2048 * 80);
    float* FF  = alloc(2048 * 320);
    float* GXF = alloc(2048 * 256);
    float* GXB = alloc(2048 * 256);
    float* OUT0 = alloc(2048 * 128);
    float* OUT1 = alloc(2048 * 128);
    (void)ws_size; (void)n_in; (void)in_sizes; (void)out_size;

    // --- per-feature LSTMs ---
    vlstm_kernel<<<(640 + 63) / 64, 64, 0, stream>>>(x, vWih, vWhh, vb, Vv);
    feat_lstm_kernel<<<5, 256, 0, stream>>>(x, qWih, qb, qWhh, Qh);
    feat_lstm_kernel<<<5, 256, 0, stream>>>(x, kWih, kb, kWhh, Kh);

    // --- feature attention: scores/softmax + b output ---
    scores_kernel<<<640, 64, 0, stream>>>(Qh, Kh, Vv, WSM, bOut);

    // --- src = [x, x @ Wsm^T] ---
    copyx_kernel<<<(2048 * 40 + 255) / 256, 256, 0, stream>>>(x, SRC);
    launch_gemm(stream, x, WSM, nullptr, SRC + 40, 128, 40, 40, 40, 40, 80,
                16, 1, 128 * 40, 0, 1600, 0, 128 * 80, 0, 1, 0, 1.0f);

    // --- 2 transformer encoder layers ---
    for (int l = 0; l < 2; ++l) {
        const float* Wq = tWq + l * 6400;  const float* bq = tbq + l * 80;
        const float* Wk = tWk + l * 6400;  const float* bk = tbk + l * 80;
        const float* Wv = tWv + l * 6400;  const float* bv = tbv + l * 80;
        const float* Wo = tWo + l * 6400;  const float* bo = tbo + l * 80;
        const float* W1 = tW1 + l * 25600; const float* b1 = tb1 + l * 320;
        const float* W2 = tW2 + l * 25600; const float* b2 = tb2 + l * 80;
        const float* g  = tlng + l * 80;   const float* bbet = tlnb + l * 80;
        float* cL = cOut + (long long)l * 16 * 5 * 128 * 128;

        shuffle_kernel<<<(2048 * 80 + 255) / 256, 256, 0, stream>>>(SRC, QS);
        launch_gemm(stream, QS, Wq, bq, QB, 2048, 80, 80, 80, 80, 80,
                    1, 1, 0, 0, 0, 0, 0, 0, 1, 0, 1.0f);
        launch_gemm(stream, QB, Wk, bk, KB, 2048, 80, 80, 80, 80, 80,
                    1, 1, 0, 0, 0, 0, 0, 0, 1, 0, 1.0f);
        launch_gemm(stream, QB, Wv, bv, VB, 2048, 80, 80, 80, 80, 80,
                    1, 1, 0, 0, 0, 0, 0, 0, 1, 0, 1.0f);
        // scores = qh @ kh^T / 4   -> directly into c output region
        launch_gemm(stream, QB, KB, nullptr, cL, 128, 128, 16, 80, 80, 128,
                    80, 5, 128 * 80, 16, 128 * 80, 16, 5 * 16384, 16384, 1, 0, 0.25f);
        softmax128_kernel<<<(10240 + 3) / 4, 128, 0, stream>>>(cL, 10240);
        // context = attn @ vh -> merged heads in OB
        launch_gemm(stream, cL, VB, nullptr, OB, 128, 16, 128, 128, 80, 80,
                    80, 5, 5 * 16384, 16384, 128 * 80, 16, 128 * 80, 16, 0, 0, 1.0f);
        launch_gemm(stream, OB, Wo, bo, QB, 2048, 80, 80, 80, 80, 80,
                    1, 1, 0, 0, 0, 0, 0, 0, 1, 0, 1.0f);
        add_ln_kernel<<<512, 128, 0, stream>>>(SRC, QB, g, bbet, 2048);
        launch_gemm(stream, SRC, W1, b1, FF, 2048, 320, 80, 80, 80, 320,
                    1, 1, 0, 0, 0, 0, 0, 0, 1, 1, 1.0f);
        launch_gemm(stream, FF, W2, b2, QB, 2048, 80, 320, 320, 320, 80,
                    1, 1, 0, 0, 0, 0, 0, 0, 1, 0, 1.0f);
        add_ln_kernel<<<512, 128, 0, stream>>>(SRC, QB, g, bbet, 2048);
    }

    // --- bi-LSTM layer 0 ---
    launch_gemm(stream, SRC, l0fWih, l0fb, GXF, 2048, 256, 80, 80, 80, 256,
                1, 1, 0, 0, 0, 0, 0, 0, 1, 0, 1.0f);
    launch_gemm(stream, SRC, l0bWih, l0bb, GXB, 2048, 256, 80, 80, 80, 256,
                1, 1, 0, 0, 0, 0, 0, 0, 1, 0, 1.0f);
    bilstm_rec_kernel<<<2, 256, 0, stream>>>(GXF, GXB, l0fWhh, l0bWhh, OUT0);

    // --- bi-LSTM layer 1 ---
    launch_gemm(stream, OUT0, l1fWih, l1fb, GXF, 2048, 256, 128, 128, 128, 256,
                1, 1, 0, 0, 0, 0, 0, 0, 1, 0, 1.0f);
    launch_gemm(stream, OUT0, l1bWih, l1bb, GXB, 2048, 256, 128, 128, 128, 256,
                1, 1, 0, 0, 0, 0, 0, 0, 1, 0, 1.0f);
    bilstm_rec_kernel<<<2, 256, 0, stream>>>(GXF, GXB, l1fWhh, l1bWhh, OUT1);

    // --- head ---
    head_kernel<<<16, 128, 0, stream>>>(OUT1, fc1W, fc1b, fc2W, fc2b, outP);
}
